// LSTM_Action_Classifier_55654186221631
// MI455X (gfx1250) — matrix operand order repaired and checked by reading, not compiled, not run on hardware
//
#include <hip/hip_runtime.h>
#include <hip/hip_bf16.h>

// ---------------------------------------------------------------------------
// Problem constants (from reference)
// ---------------------------------------------------------------------------
#define BB   64
#define TT   2048
#define HH   128
#define DIN  108
#define LL   37
#define TB   ((size_t)TT * BB)       // 131072 rows (t-major: row = t*B + b)
#define KP   128                     // padded K for layer-0 input (108 -> 128)
#define FOUR_H 512

typedef __attribute__((ext_vector_type(16))) __bf16 v16bf;
typedef __attribute__((ext_vector_type(8)))  float  v8f;

union ABf  { v16bf v; unsigned int u[8]; __bf16 h[16]; };
union U4BF { uint4 q; __bf16 h[8]; };

// fast activations on the TRANS pipe (v_exp_f32 + v_rcp_f32)
__device__ inline float fast_sig(float x) {
  return __builtin_amdgcn_rcpf(1.f + __expf(-x));
}
__device__ inline float fast_tanh(float x) {
  return 2.f * __builtin_amdgcn_rcpf(1.f + __expf(-2.f * x)) - 1.f;
}

// ---------------------------------------------------------------------------
// WMMA fragment loaders (layouts per CDNA5 ISA 7.12.2, wave32)
// A: 16x32 bf16 from row-major source (ld elements):
//   lanes 0-15 : V0..3 -> K=0..7,  V4..7 -> K=16..23   (row = lane)
//   lanes16-31 : V0..3 -> K=8..15, V4..7 -> K=24..31
// ---------------------------------------------------------------------------
__device__ inline v16bf load_a_16x32(const __bf16* __restrict__ base, int ld) {
  int lane = threadIdx.x & 31;
  int half = lane >> 4;
  int r    = lane & 15;
  const __bf16* rowp = base + (size_t)r * ld;
  ABf x;
#pragma unroll
  for (int j = 0; j < 8; ++j) {
    int kb = ((j & 4) ? 16 : 0) + half * 8 + (j & 3) * 2;
    x.u[j] = *(const unsigned int*)(rowp + kb);            // pair of bf16
  }
  return x.v;
}

// A fragment from a column-major source: element (r,k) at base[k*ldc + r].
// Scalar b16 gathers, but consecutive lanes hit consecutive rows -> coalesced.
__device__ inline v16bf load_a_16x32_cm(const __bf16* __restrict__ base, size_t ldc) {
  int lane = threadIdx.x & 31;
  int half = lane >> 4;
  int r    = lane & 15;
  ABf x;
#pragma unroll
  for (int j = 0; j < 8; ++j) {
    int kb = ((j & 4) ? 16 : 0) + half * 8 + (j & 3) * 2;
    x.h[2 * j]     = base[(size_t)kb * ldc + r];
    x.h[2 * j + 1] = base[(size_t)(kb + 1) * ldc + r];
  }
  return x.v;
}

// B: 32x16 bf16 where B[k][n] = W[n][k], W row-major [.,ldw]:
//   lanes 0-15 : V0..3 -> K=0..7,  V4..7 -> K=16..23  (col n = lane)
//   lanes16-31 : V0..3 -> K=8..15, V4..7 -> K=24..31
__device__ inline v16bf load_b_32x16(const __bf16* __restrict__ w, int ldw,
                                     int n0, int k0) {
  int lane = threadIdx.x & 31;
  int kh = lane >> 4;
  int n  = lane & 15;
  const __bf16* rowp = w + (size_t)(n0 + n) * ldw + k0 + kh * 8;
  ABf x;
#pragma unroll
  for (int j = 0; j < 8; ++j)
    x.u[j] = *(const unsigned int*)(rowp + 2 * j + ((j & 4) ? 8 : 0));
  return x.v;
}

// C/D 16x16 f32: VGPR j -> row m = j + 8*(lane>>4), col n = lane&15.

// ---------------------------------------------------------------------------
// Pack inputs: X[t*B+b][0..127] = [x_hand*mask (99) | x_obj (9) | zeros (20)]
// ---------------------------------------------------------------------------
__global__ void k_pack_x(const float* __restrict__ xh, const float* __restrict__ xo,
                         const int* __restrict__ is_hand, __bf16* __restrict__ X) {
  size_t i = (size_t)blockIdx.x * blockDim.x + threadIdx.x;   // over TB*128
  if (i >= TB * (size_t)KP) return;
  int    k = (int)(i & (KP - 1));
  size_t r = i >> 7;                    // t*B + b
  int    b = (int)(r & (BB - 1));
  size_t t = r >> 6;                    // B = 64
  float v = 0.f;
  if (k < 99)       v = xh[((size_t)b * TT + t) * 99 + k] * (is_hand[b] ? 1.f : 0.f);
  else if (k < DIN) v = xo[((size_t)b * TT + t) * 9 + (k - 99)];
  X[i] = (__bf16)v;
}

// Convert a [512, scols] f32 weight into padded [512, 128] bf16
__global__ void k_cvt_pad(const float* __restrict__ src, __bf16* __restrict__ dst,
                          int scols) {
  int i = blockIdx.x * blockDim.x + threadIdx.x;
  if (i >= FOUR_H * KP) return;
  int r = i >> 7, c = i & (KP - 1);
  dst[i] = (__bf16)((c < scols) ? src[r * scols + c] : 0.f);
}

__global__ void k_add2(const float* __restrict__ a, const float* __restrict__ b,
                       float* __restrict__ d) {
  int i = blockIdx.x * blockDim.x + threadIdx.x;
  if (i < FOUR_H) d[i] = a[i] + b[i];
}

// ---------------------------------------------------------------------------
// Batched input-projection GEMM (register-blocked):
//   Gcm[col][row] (bf16, col-major, col stride = M) = A[M,128] @ W[512,128]^T
// One wave computes 64 rows x 512 cols. kk is the outer MAC loop so each B
// fragment feeds 4 INDEPENDENT accumulators (no WMMA RAW chains back-to-back).
// ACM=true reads A from a column-major image (ldA = row stride).
// ---------------------------------------------------------------------------
template <bool ACM>
__global__ __launch_bounds__(256)
void k_gemm_bf16(const __bf16* __restrict__ A, const __bf16* __restrict__ W,
                 __bf16* __restrict__ Gcm, int M, size_t ldA) {
  int wave = (int)((blockIdx.x * blockDim.x + threadIdx.x) >> 5);
  int m0 = wave * 64;
  if (m0 >= M) return;                     // wave-uniform

  v16bf a[4][4];                           // [m-tile][k-tile]
#pragma unroll
  for (int mt = 0; mt < 4; ++mt)
#pragma unroll
    for (int kk = 0; kk < 4; ++kk)
      a[mt][kk] = ACM
          ? load_a_16x32_cm(A + (size_t)(kk * 32) * ldA + m0 + mt * 16, ldA)
          : load_a_16x32(A + (size_t)(m0 + mt * 16) * KP + kk * 32, KP);

  int lane = threadIdx.x & 31;
  int half = lane >> 4, n = lane & 15;

  for (int nt = 0; nt < 32; ++nt) {
    v8f acc[4] = {};
#pragma unroll
    for (int kk = 0; kk < 4; ++kk) {
      v16bf b = load_b_32x16(W, KP, nt * 16, kk * 32);
#pragma unroll
      for (int mt = 0; mt < 4; ++mt)
        acc[mt] = __builtin_amdgcn_wmma_f32_16x16x32_bf16(
            false, a[mt][kk], false, b, (short)0, acc[mt], false, false);
    }
#pragma unroll
    for (int mt = 0; mt < 4; ++mt) {
      U4BF st;
#pragma unroll
      for (int j = 0; j < 8; ++j) st.h[j] = (__bf16)acc[mt][j];
      // column-major: 8 consecutive rows -> one b128 store
      *(uint4*)(Gcm + (size_t)(nt * 16 + n) * (size_t)M + m0 + mt * 16 + 8 * half) =
          st.q;
    }
  }
}

// ---------------------------------------------------------------------------
// Persistent recurrent LSTM layer. grid = 2 (branch), 512 threads (16 waves).
// Whh (512x128 bf16) resident in LDS; h DOUBLE-BUFFERED in LDS (one barrier
// per step); c lives in VGPRs. Wave (mt, ng): rows b in [16*mt,16*mt+16),
// hidden j in [32*ng, 32*ng+32), owning ALL FOUR gate tiles -> pointwise
// update entirely in registers. Gx / H1out are column-major (stride TB).
// ---------------------------------------------------------------------------
__global__ __launch_bounds__(512, 1)
void k_lstm_rec(const __bf16* __restrict__ Gx,    // [2] cm [512][TB] (no bias)
                const __bf16* __restrict__ Whh,   // [2][512*128] bf16
                const float*  __restrict__ bias,  // [2][512] f32 (bih+bhh)
                const int*    __restrict__ duration,
                __bf16* __restrict__ H1out,       // layer0: [2] cm [128][TB]; else null
                float*  __restrict__ hfout) {     // layer1: [2][64*128];  else null
  extern __shared__ char smem[];
  __bf16* sW  = (__bf16*)smem;                              // 512*128 = 128 KB
  __bf16* sH0 = (__bf16*)(smem + FOUR_H * KP * 2);          // 64*128  =  16 KB
  __bf16* sH1 = sH0 + BB * HH;                              // 64*128  =  16 KB

  int br = blockIdx.x;
  Gx   += (size_t)br * TB * FOUR_H;
  Whh  += (size_t)br * FOUR_H * KP;
  bias += br * FOUR_H;
  if (H1out) H1out += (size_t)br * TB * HH;
  if (hfout) hfout += (size_t)br * BB * HH;

  int tid = threadIdx.x;
  // stage weights into LDS, zero h ping-pong buffers
  {
    const unsigned int* wsrc = (const unsigned int*)Whh;
    unsigned int* wdst = (unsigned int*)sW;
    for (int i = tid; i < FOUR_H * KP / 2; i += 512) wdst[i] = wsrc[i];
    unsigned int* hz = (unsigned int*)sH0;
    for (int i = tid; i < BB * HH; i += 512) hz[i] = 0u;    // both buffers
  }
  __syncthreads();

  int wv = tid >> 5, lane = tid & 31;
  int mt = wv & 3, ng = wv >> 2;          // b-tile, j-group
  int half = lane >> 4, nn = lane & 15;
  int brow = 16 * mt + 8 * half;          // + j gives batch row

  float bq[4][2];
#pragma unroll
  for (int q = 0; q < 4; ++q)
#pragma unroll
    for (int e = 0; e < 2; ++e)
      bq[q][e] = bias[128 * q + 32 * ng + 16 * e + nn];

  int durj[8];
#pragma unroll
  for (int j = 0; j < 8; ++j) durj[j] = duration[brow + j];

  v8f c_e0 = {}, c_e1 = {};
  const int PD = 8;                        // prefetch distance (one 128B line/step)

  for (int t = 0; t < TT; ++t) {
    const __bf16* sHr = (t & 1) ? sH1 : sH0;   // read h(t)
    __bf16*       sHw = (t & 1) ? sH0 : sH1;   // write h(t+1)

    // gate pre-activations for this step: 8 packed b128 loads, issued early so
    // global latency overlaps the WMMA block below
    U4BF gq[4][2];
#pragma unroll
    for (int q = 0; q < 4; ++q)
#pragma unroll
      for (int e = 0; e < 2; ++e) {
        int col = 128 * q + 32 * ng + 16 * e + nn;
        gq[q][e].q = *(const uint4*)(Gx + (size_t)col * TB + (size_t)t * BB + brow);
      }
    if (t + PD < TT) {
#pragma unroll
      for (int q = 0; q < 4; ++q)
#pragma unroll
        for (int e = 0; e < 2; ++e) {
          int col = 128 * q + 32 * ng + 16 * e + nn;
          __builtin_prefetch(Gx + (size_t)col * TB + (size_t)(t + PD) * BB + brow,
                             0, 1);
        }
    }

    // read previous h from the read buffer (never written this step)
    v16bf a[4];
#pragma unroll
    for (int kk = 0; kk < 4; ++kk)
      a[kk] = load_a_16x32(sHr + (size_t)(16 * mt) * HH + kk * 32, HH);

    // recurrent GEMM, kk-outer: 8 independent accumulator chains interleave,
    // so consecutive WMMAs share no registers (no RAW stalls / hazard NOPs)
    v8f acc[4][2] = {};
#pragma unroll
    for (int kk = 0; kk < 4; ++kk)
#pragma unroll
      for (int q = 0; q < 4; ++q)
#pragma unroll
        for (int e = 0; e < 2; ++e) {
          v16bf b = load_b_32x16(sW, KP, (8 * q + 2 * ng + e) * 16, kk * 32);
          acc[q][e] = __builtin_amdgcn_wmma_f32_16x16x32_bf16(
              false, a[kk], false, b, (short)0, acc[q][e], false, false);
        }

    // pointwise LSTM update, entirely in registers (gate order i,f,g,o)
#pragma unroll
    for (int e = 0; e < 2; ++e) {
      int jcol = 32 * ng + 16 * e + nn;
      U4BF hst;
#pragma unroll
      for (int j = 0; j < 8; ++j) {
        int bb = brow + j;
        float gi = (float)gq[0][e].h[j] + acc[0][e][j] + bq[0][e];
        float gf = (float)gq[1][e].h[j] + acc[1][e][j] + bq[1][e];
        float gg = (float)gq[2][e].h[j] + acc[2][e][j] + bq[2][e];
        float go = (float)gq[3][e].h[j] + acc[3][e][j] + bq[3][e];
        float cprev = e ? c_e1[j] : c_e0[j];
        float cv = fast_sig(gf) * cprev + fast_sig(gi) * fast_tanh(gg);
        if (e) c_e1[j] = cv; else c_e0[j] = cv;
        float hv = fast_sig(go) * fast_tanh(cv);
        __bf16 hb = (__bf16)hv;
        hst.h[j] = hb;
        sHw[(size_t)bb * HH + jcol] = hb;
        if (hfout && t == durj[j] - 1) hfout[(size_t)bb * HH + jcol] = hv;
      }
      if (H1out)   // column-major: 8 consecutive rows -> one b128 store
        *(uint4*)(H1out + (size_t)jcol * TB + (size_t)t * BB + brow) = hst.q;
    }
    __syncthreads();   // single barrier per step (double-buffered h)
  }
}

// ---------------------------------------------------------------------------
// Epilogue: conv1d(k=3) over features, branch sum, logits, softmax, BCE, top-k
// out layout: [loss(1) | top1(64) | top3(64) | act(64*128) | pred(64*37)]
// ---------------------------------------------------------------------------
__global__ __launch_bounds__(128)
void k_final(const float* __restrict__ hf,   // [2][64*128]
             const float* __restrict__ wcl, const float* __restrict__ bcl,
             const float* __restrict__ wcr, const float* __restrict__ bcr,
             const float* __restrict__ Wout, const float* __restrict__ bout,
             const int* __restrict__ y,
             float* __restrict__ out, float* __restrict__ lossp) {
  int b = blockIdx.x, j = threadIdx.x;
  __shared__ float hl[HH], hr[HH], y3[HH], lg[LL], p[LL];
  hl[j] = hf[b * HH + j];
  hr[j] = hf[BB * HH + b * HH + j];
  __syncthreads();
  float l0 = (j > 0) ? hl[j - 1] : 0.f, l2 = (j < HH - 1) ? hl[j + 1] : 0.f;
  float r0 = (j > 0) ? hr[j - 1] : 0.f, r2 = (j < HH - 1) ? hr[j + 1] : 0.f;
  float v = wcl[0] * l0 + wcl[1] * hl[j] + wcl[2] * l2 + bcl[0]
          + wcr[0] * r0 + wcr[1] * hr[j] + wcr[2] * r2 + bcr[0];
  y3[j] = v;
  out[1 + 64 + 64 + b * HH + j] = tanhf(v) * 0.1f;     // activation
  __syncthreads();
  if (j < LL) {
    float s = bout[j];
    for (int k = 0; k < HH; ++k) s += Wout[j * HH + k] * y3[k];
    lg[j] = s;
  }
  __syncthreads();
  if (j == 0) {
    float mx = -1e30f;
    for (int l = 0; l < LL; ++l) mx = fmaxf(mx, lg[l]);
    float den = 0.f;
    for (int l = 0; l < LL; ++l) { p[l] = __expf(lg[l] - mx); den += p[l]; }
    for (int l = 0; l < LL; ++l) p[l] /= den;
    int tgt = y[b];
    float pt = p[tgt];
    int rank = 0;
    float lsum = 0.f;
    for (int l = 0; l < LL; ++l) {
      out[1 + 64 + 64 + BB * HH + b * LL + l] = p[l];
      float lp = fmaxf(logf(p[l]), -100.f);
      float lq = fmaxf(logf(1.f - p[l]), -100.f);
      float tt = (l == tgt) ? 1.f : 0.f;
      lsum += tt * lp + (1.f - tt) * lq;
      if (p[l] > pt || (p[l] == pt && l < tgt)) rank++;
    }
    out[1 + b]      = (rank < 1) ? 1.f : 0.f;
    out[1 + 64 + b] = (rank < 3) ? 1.f : 0.f;
    lossp[b] = lsum;
  }
}

__global__ void k_reduce(const float* __restrict__ lp, float* __restrict__ out) {
  if (blockIdx.x == 0 && threadIdx.x == 0) {
    float s = 0.f;
    for (int i = 0; i < BB; ++i) s += lp[i];      // deterministic order
    out[0] = -s / (float)(BB * LL);
  }
}

// ---------------------------------------------------------------------------
// Host launcher
// ---------------------------------------------------------------------------
extern "C" void kernel_launch(void* const* d_in, const int* in_sizes, int n_in,
                              void* d_out, int out_size, void* d_ws, size_t ws_size,
                              hipStream_t stream) {
  (void)in_sizes; (void)n_in; (void)out_size; (void)ws_size;

  const float* x_l  = (const float*)d_in[0];
  const float* x_r  = (const float*)d_in[1];
  const float* x_o  = (const float*)d_in[2];
  const int*   y    = (const int*)d_in[3];
  const int*   is_l = (const int*)d_in[4];
  const int*   is_r = (const int*)d_in[5];
  const int*   dur  = (const int*)d_in[6];
  const float* W[16];
  for (int i = 0; i < 16; ++i) W[i] = (const float*)d_in[7 + i];
  const float* wcl  = (const float*)d_in[23];
  const float* bcl  = (const float*)d_in[24];
  const float* wcr  = (const float*)d_in[25];
  const float* bcr  = (const float*)d_in[26];
  const float* Wout = (const float*)d_in[27];
  const float* bout = (const float*)d_in[28];

  // workspace layout
  const size_t S = (size_t)FOUR_H * KP;                  // one weight matrix
  char* ws = (char*)d_ws;
  size_t off = 0;
  __bf16* Xbf = (__bf16*)(ws + off); off += 2 * TB * (size_t)KP * 2;
  __bf16* G   = (__bf16*)(ws + off); off += 2 * TB * (size_t)FOUR_H * 2;
  __bf16* H1  = (__bf16*)(ws + off); off += 2 * TB * (size_t)HH * 2;
  __bf16* WB  = (__bf16*)(ws + off); off += 8 * S * 2;
  float*  BIAS = (float*)(ws + off); off += 4 * FOUR_H * 4;
  float*  HF   = (float*)(ws + off); off += 2 * BB * HH * 4;
  float*  LOSSP = (float*)(ws + off);

  const size_t smem_rec = (size_t)FOUR_H * KP * 2 + 2 * (size_t)BB * HH * 2; // 163840
  hipFuncSetAttribute((const void*)k_lstm_rec,
                      hipFuncAttributeMaxDynamicSharedMemorySize, (int)smem_rec);

  // 1) pack inputs to bf16 (t-major rows)
  {
    size_t nx = TB * (size_t)KP;
    int blocks = (int)((nx + 255) / 256);
    k_pack_x<<<blocks, 256, 0, stream>>>(x_l, x_o, is_l, Xbf);
    k_pack_x<<<blocks, 256, 0, stream>>>(x_r, x_o, is_r, Xbf + nx);
  }
  // 2) convert weights (padded) + fuse biases
  {
    int blocks = (FOUR_H * KP + 255) / 256;
    // layout: [Wih0 l,r | Whh0 l,r | Wih1 l,r | Whh1 l,r]
    k_cvt_pad<<<blocks, 256, 0, stream>>>(W[0],  WB + 0 * S, DIN); // Wih0_l
    k_cvt_pad<<<blocks, 256, 0, stream>>>(W[8],  WB + 1 * S, DIN); // Wih0_r
    k_cvt_pad<<<blocks, 256, 0, stream>>>(W[1],  WB + 2 * S, HH);  // Whh0_l
    k_cvt_pad<<<blocks, 256, 0, stream>>>(W[9],  WB + 3 * S, HH);  // Whh0_r
    k_cvt_pad<<<blocks, 256, 0, stream>>>(W[4],  WB + 4 * S, HH);  // Wih1_l
    k_cvt_pad<<<blocks, 256, 0, stream>>>(W[12], WB + 5 * S, HH);  // Wih1_r
    k_cvt_pad<<<blocks, 256, 0, stream>>>(W[5],  WB + 6 * S, HH);  // Whh1_l
    k_cvt_pad<<<blocks, 256, 0, stream>>>(W[13], WB + 7 * S, HH);  // Whh1_r
    k_add2<<<2, 256, 0, stream>>>(W[2],  W[3],  BIAS + 0 * FOUR_H); // b0 l
    k_add2<<<2, 256, 0, stream>>>(W[10], W[11], BIAS + 1 * FOUR_H); // b0 r
    k_add2<<<2, 256, 0, stream>>>(W[6],  W[7],  BIAS + 2 * FOUR_H); // b1 l
    k_add2<<<2, 256, 0, stream>>>(W[14], W[15], BIAS + 3 * FOUR_H); // b1 r
  }

  const int gemm_blocks = (int)((TB / 64 + 7) / 8);   // 8 waves/block, 64 rows/wave

  // 3) layer-0 input projection (both branches), then recurrent layer 0
  for (int br = 0; br < 2; ++br)
    k_gemm_bf16<false><<<gemm_blocks, 256, 0, stream>>>(
        Xbf + (size_t)br * TB * KP, WB + (size_t)br * S,
        G + (size_t)br * TB * FOUR_H, (int)TB, (size_t)KP);
  k_lstm_rec<<<2, 512, smem_rec, stream>>>(G, WB + 2 * S, BIAS + 0, dur,
                                           H1, (float*)nullptr);

  // 4) layer-1 input projection (A = column-major H1), then recurrent layer 1
  for (int br = 0; br < 2; ++br)
    k_gemm_bf16<true><<<gemm_blocks, 256, 0, stream>>>(
        H1 + (size_t)br * TB * HH, WB + 4 * S + (size_t)br * S,
        G + (size_t)br * TB * FOUR_H, (int)TB, TB);
  k_lstm_rec<<<2, 512, smem_rec, stream>>>(G, WB + 6 * S, BIAS + 2 * FOUR_H, dur,
                                           (__bf16*)nullptr, HF);

  // 5) epilogue
  k_final<<<BB, HH, 0, stream>>>(HF, wcl, bcl, wcr, bcr, Wout, bout, y,
                                 (float*)d_out, LOSSP);
  k_reduce<<<1, 32, 0, stream>>>(LOSSP, (float*)d_out);
}
